// Assign_62766652064354
// MI455X (gfx1250) — matrix operand order repaired and checked by reading, not compiled, not run on hardware
//
#include <hip/hip_runtime.h>

// -------- problem constants --------
#define NC    156          // clusters
#define NCP   160          // padded to 10 tiles of 16
#define KT    32           // pixels per WMMA K-step
#define STP   40           // LDS s-tile row stride in halves (80B: 16B aligned, 20-bank stride)
#define NW    10           // waves per block (tile rows)
#define TPB   (NW * 32)    // 320 threads
#define STEPS 32           // K-steps per block: one full image row (32*32 = 1024 = W)
#define NBLK  1024         // one block per image row
#define HW    1024
#define NTAP  18           // 3*3*2 conv taps

typedef __attribute__((ext_vector_type(16))) _Float16 v16h;
typedef __attribute__((ext_vector_type(8)))  _Float16 v8h;
typedef __attribute__((ext_vector_type(8)))  float    v8f;

// ---------------- prologue: zero adj output + workspace accumulators ----------------
__global__ void zero_init_kernel(float* __restrict__ out, int n_out,
                                 float* __restrict__ ws, int n_ws) {
    int i = blockIdx.x * blockDim.x + threadIdx.x;
    if (i < n_out) out[i] = 0.0f;
    if (i < n_ws)  ws[i]  = 0.0f;
}

// ---------------- fused conv3x3 + softmax + hard-assign + WMMA s^T s ----------------
__global__ __launch_bounds__(TPB) void fused_assign_kernel(
    const float* __restrict__ in,   // [1024,1024,2] NHWC
    const float* __restrict__ cw,   // [3,3,2,156] HWIO
    const float* __restrict__ cb,   // [156]
    float* __restrict__ out,        // [312 nodes | 24336 adj]
    float* __restrict__ ws)         // [156 cnt | 156 sx | 156 sy]
{
    __shared__ float2 rows[3][HW];                       // staged input rows y-1..y+1 (zero-padded)
    __shared__ __align__(16) float WL[NTAP * NC + 16];   // staged conv weights (+pad for clamped reads)
    __shared__ __align__(16) float BL[NCP];              // staged bias (+pad)
    __shared__ __align__(16) _Float16 ST[NCP][STP];      // s tile, channel-major f16
    __shared__ float redM[NW][KT];                       // per-pixel max partials
    __shared__ float redS[NW][KT];                       // per-pixel sum partials
    __shared__ float lcnt[NCP], lsx[NCP], lsy[NCP];      // block-local hard-assign accum

    const int t    = threadIdx.x;
    const int lane = t & 31;
    const int wid  = t >> 5;        // 0..9  (wave id == channel-tile / adj tile-row)
    const int c0   = wid << 4;      // channel group base
    const int y    = blockIdx.x;    // this block processes image row y

    // ---- one-time staging (branch-free: clamp + multiplicative mask) ----
    const float2* in2 = (const float2*)in;
    for (int idx = t; idx < 3 * HW; idx += TPB) {
        int ry = idx >> 10, rx = idx & (HW - 1);
        int yy = y + ry - 1;
        float m  = (yy >= 0 && yy < HW) ? 1.0f : 0.0f;
        int   yc = min(max(yy, 0), HW - 1);
        float2 v = in2[(yc << 10) + rx];
        rows[ry][rx] = make_float2(v.x * m, v.y * m);
    }
    for (int idx = t; idx < NTAP * NC + 16; idx += TPB)
        WL[idx] = (idx < NTAP * NC) ? cw[idx] : 0.0f;
    for (int idx = t; idx < NCP; idx += TPB) {
        BL[idx] = (idx < NC) ? cb[idx] : 0.0f;
        lcnt[idx] = 0.f; lsx[idx] = 0.f; lsy[idx] = 0.f;
    }
    __syncthreads();

    v8f acc[NW];
    #pragma unroll
    for (int j = 0; j < NW; ++j)
        #pragma unroll
        for (int r = 0; r < 8; ++r) acc[j][r] = 0.0f;

    for (int step = 0; step < STEPS; ++step) {
        // ---- conv 3x3 for pixel x = step*32+lane, channels [c0, c0+16) ----
        const int x = (step << 5) + lane;

        float mx[3]; int xc[3];
        #pragma unroll
        for (int dx = 0; dx < 3; ++dx) {
            int xx = x + dx - 1;
            mx[dx] = (xx >= 0 && xx < HW) ? 1.0f : 0.0f;
            xc[dx] = min(max(xx, 0), HW - 1);
        }
        float iv[NTAP];
        #pragma unroll
        for (int dy = 0; dy < 3; ++dy)
            #pragma unroll
            for (int dx = 0; dx < 3; ++dx) {
                float2 v = rows[dy][xc[dx]];
                iv[(dy * 3 + dx) * 2 + 0] = v.x * mx[dx];
                iv[(dy * 3 + dx) * 2 + 1] = v.y * mx[dx];
            }

        // bias init: 4x ds_load_b128 (wave-uniform broadcast)
        float val[16];
        #pragma unroll
        for (int q = 0; q < 4; ++q) {
            float4 b = ((const float4*)&BL[c0])[q];
            val[4 * q + 0] = b.x; val[4 * q + 1] = b.y;
            val[4 * q + 2] = b.z; val[4 * q + 3] = b.w;
        }
        // conv accumulate: per tap, 4x ds_load_b128 of 16 consecutive channel weights
        #pragma unroll
        for (int tap = 0; tap < NTAP; ++tap) {
            const float4* w4 = (const float4*)&WL[tap * NC + c0];
            float ivt = iv[tap];
            #pragma unroll
            for (int q = 0; q < 4; ++q) {
                float4 w = w4[q];
                val[4 * q + 0] = fmaf(ivt, w.x, val[4 * q + 0]);
                val[4 * q + 1] = fmaf(ivt, w.y, val[4 * q + 1]);
                val[4 * q + 2] = fmaf(ivt, w.z, val[4 * q + 2]);
                val[4 * q + 3] = fmaf(ivt, w.w, val[4 * q + 3]);
            }
        }
        // mask pad channels (wave 9, cc>=12) -> -inf so exp==0 downstream
        #pragma unroll
        for (int cc = 0; cc < 16; ++cc)
            if ((c0 + cc) >= NC) val[cc] = -3.0e38f;

        // ---- softmax across 10 waves (max, then exp-sum) ----
        float m = val[0];
        #pragma unroll
        for (int cc = 1; cc < 16; ++cc) m = fmaxf(m, val[cc]);
        redM[wid][lane] = m;
        __syncthreads();
        float pm = redM[0][lane];
        #pragma unroll
        for (int w2 = 1; w2 < NW; ++w2) pm = fmaxf(pm, redM[w2][lane]);

        float ev[16];
        float ls = 0.0f;
        #pragma unroll
        for (int cc = 0; cc < 16; ++cc) {
            float e = __expf(val[cc] - pm);      // pad channels: exp(-huge) == 0
            ev[cc] = e; ls += e;
        }
        redS[wid][lane] = ls;
        __syncthreads();
        float Z = 0.0f;
        #pragma unroll
        for (int w2 = 0; w2 < NW; ++w2) Z += redS[w2][lane];
        const float invZ = 1.0f / Z;

        // ---- store s tile (f16, channel-major) + hard assignment ----
        const float cx = iv[8], cy = iv[9];      // center pixel coords (dy=dx=1)
        #pragma unroll
        for (int cc = 0; cc < 16; ++cc) {
            float s = ev[cc] * invZ;
            ST[c0 + cc][lane] = (_Float16)s;
            if (s > 0.5f) {                      // pad channels give s==0, never taken
                atomicAdd(&lcnt[c0 + cc], 1.0f);
                atomicAdd(&lsx[c0 + cc], cx);
                atomicAdd(&lsy[c0 + cc], cy);
            }
        }
        __syncthreads();   // s tile ready; EXEC all-ones for WMMA below

        // ---- WMMA rank-32 update: wave `wid` owns adj tile-row `wid` ----
        const int mcol = lane & 15;
        const int kA   = (lane >> 4) << 4;   // 0 / 16
        const int kB   = (lane >> 4) << 4;   // 0 / 16

        // A (16x32): A[m][k] = s[pixel k][channel c0+m]
        v8h alo = *(const v8h*)&ST[c0 + mcol][kA];
        v8h ahi = *(const v8h*)&ST[c0 + mcol][kA + 8];
        v16h A = __builtin_shufflevector(alo, ahi,
                    0,1,2,3,4,5,6,7,8,9,10,11,12,13,14,15);

        // B (32x16): B[k][n] = s[pixel k][channel 16j+n]
        // one-deep software pipeline: issue loads for tile j+1 before consuming tile j
        v8h blo = *(const v8h*)&ST[mcol][kB];
        v8h bhi = *(const v8h*)&ST[mcol][kB + 8];
        #pragma unroll
        for (int j = 0; j < NW; ++j) {
            v16h B = __builtin_shufflevector(blo, bhi,
                        0,1,2,3,4,5,6,7,8,9,10,11,12,13,14,15);
            if (j < NW - 1) {
                blo = *(const v8h*)&ST[((j + 1) << 4) + mcol][kB];
                bhi = *(const v8h*)&ST[((j + 1) << 4) + mcol][kB + 8];
            }
            acc[j] = __builtin_amdgcn_wmma_f32_16x16x32_f16(
                         false, A, false, B, (short)0, acc[j], false, false);
        }
        // no barrier needed here: next step's first ST write is ordered behind
        // two more barriers (redM-sync, redS-sync) after every thread's WMMA reads
    }

    // ---- flush adjacency accumulators (D layout: row = r + 8*(lane>=16), col = lane%16) ----
    float* adj = out + 2 * NC;
    const int mcol  = lane & 15;
    const int rbase = (lane >> 4) << 3;
    #pragma unroll
    for (int j = 0; j < NW; ++j) {
        int col = (j << 4) + mcol;
        #pragma unroll
        for (int r = 0; r < 8; ++r) {
            int row = c0 + rbase + r;
            if (row < NC && col < NC)
                atomicAdd(&adj[row * NC + col], acc[j][r]);
        }
    }

    __syncthreads();
    for (int i = t; i < NC; i += TPB) {
        atomicAdd(&ws[i],          lcnt[i]);
        atomicAdd(&ws[NC + i],     lsx[i]);
        atomicAdd(&ws[2 * NC + i], lsy[i]);
    }
}

// ---------------- epilogue: new_nodes = sums / counts (0/0 -> NaN, as reference) ----------------
__global__ void finalize_nodes_kernel(const float* __restrict__ ws,
                                      float* __restrict__ out) {
    int c = blockIdx.x * blockDim.x + threadIdx.x;
    if (c < NC) {
        float cnt = ws[c];
        out[2 * c + 0] = ws[NC + c]     / cnt;
        out[2 * c + 1] = ws[2 * NC + c] / cnt;
    }
}

extern "C" void kernel_launch(void* const* d_in, const int* in_sizes, int n_in,
                              void* d_out, int out_size, void* d_ws, size_t ws_size,
                              hipStream_t stream) {
    const float* in = (const float*)d_in[0];   // [1,1024,1024,2]
    const float* cw = (const float*)d_in[1];   // [3,3,2,156]
    const float* cb = (const float*)d_in[2];   // [156]
    float* out = (float*)d_out;                // 312 + 24336 floats
    float* ws  = (float*)d_ws;                 // 3*156 floats used

    const int n_out = 2 * NC + NC * NC;        // 24648
    const int n_ws  = 3 * NC;                  // 468
    zero_init_kernel<<<(n_out + 255) / 256, 256, 0, stream>>>(out, n_out, ws, n_ws);
    fused_assign_kernel<<<NBLK, TPB, 0, stream>>>(in, cw, cb, out, ws);
    finalize_nodes_kernel<<<1, 256, 0, stream>>>(ws, out);
}
